// ProximalInteraction_1803886265795
// MI455X (gfx1250) — compile-verified, matrix operand order checked by reading, not run on hardware
//
#include <hip/hip_runtime.h>
#include <math.h>

typedef _Float16 v8h  __attribute__((ext_vector_type(8)));
typedef _Float16 v16h __attribute__((ext_vector_type(16)));
typedef float    v8f  __attribute__((ext_vector_type(8)));

#define BB   4
#define NN   2048
#define FF   253
#define CC   256
#define GG   256
#define UU   128
#define HH   256
#define KCMB 640          // H + C + U
#define R2   64.0f        // RADIUS^2

// ---- workspace layout (bytes) ----
#define OFF_POS4 0u          // BB*NN float4            = 131072
#define OFF_WMSG 131072u     // 256*256 f16             = 131072
#define OFF_WOUT 262144u     // 256*640 f16             = 327680
#define OFF_AGG  589824u     // BB*256 f32              = 4096
#define OFF_GUPD 593920u     // BB*128 f32              = 2048
#define OFF_H    596224u     // 8192*256 f16            = 4194304
#define OFF_COMB 4790528u    // 8192*640 f16            = 10485760
// total ~15.3 MB

// ---- d_out layout (floats) ----
#define OUT_POS   0          // [B,3,N]  = 24576
#define OUT_FEAT  24576      // [B,F,N]  = 2072576
#define OUT_GLOB  2097152    // [B,G]    = 1024

// ============================================================
// 0) pack x = concat(pos, feat) as f16 into comb[:,256:512]; build pos4
// ============================================================
__global__ void k_pack(const float* __restrict__ positions,
                       const float* __restrict__ features,
                       _Float16* __restrict__ comb,
                       float4* __restrict__ pos4) {
  int row = blockIdx.x;            // b*NN + n
  int b = row >> 11, n = row & (NN - 1);
  int c = threadIdx.x;
  float v = (c < 3) ? positions[((size_t)b * 3 + c) * NN + n]
                    : features[((size_t)b * FF + (c - 3)) * NN + n];
  comb[(size_t)row * KCMB + 256 + c] = (_Float16)v;
  if (c == 0) {
    float px = positions[((size_t)b * 3 + 0) * NN + n];
    float py = positions[((size_t)b * 3 + 1) * NN + n];
    float pz = positions[((size_t)b * 3 + 2) * NN + n];
    pos4[row] = make_float4(px, py, pz, 0.0f);
  }
}

// ============================================================
// 1) aggregated[b][c] = max over n of inputs[b,c,n]
// ============================================================
__global__ void k_maxpool(const float* __restrict__ positions,
                          const float* __restrict__ features,
                          float* __restrict__ agg) {
  __shared__ float red[256];
  int bid = blockIdx.x;            // b*CC + c
  int b = bid >> 8, c = bid & 255;
  int t = threadIdx.x;
  const float* src = (c < 3) ? positions + ((size_t)b * 3 + c) * NN
                             : features + ((size_t)b * FF + (c - 3)) * NN;
  float m = -3.4e38f;
  for (int n = t; n < NN; n += 256) m = fmaxf(m, src[n]);
  red[t] = m;
  __syncthreads();
  for (int s = 128; s > 0; s >>= 1) {
    if (t < s) red[t] = fmaxf(red[t], red[t + s]);
    __syncthreads();
  }
  if (t == 0) agg[bid] = red[0];
}

// ============================================================
// 2) convert W_msg, W_out to f16
// ============================================================
__global__ void k_cvtw(const float* __restrict__ Wmsg, const float* __restrict__ Wout,
                       _Float16* __restrict__ Wmsg_h, _Float16* __restrict__ Wout_h) {
  int idx = blockIdx.x * 256 + threadIdx.x;  // 0 .. 229375
  if (idx < HH * CC) Wmsg_h[idx] = (_Float16)Wmsg[idx];
  else               Wout_h[idx - HH * CC] = (_Float16)Wout[idx - HH * CC];
}

// ============================================================
// 3) global MLP: global_new = tanh([agg,gf] @ W_g.T + b_g); gupd = gnew @ W_u.T + b_u
// ============================================================
__global__ void k_global(const float* __restrict__ agg, const float* __restrict__ gf,
                         const float* __restrict__ Wg, const float* __restrict__ bg,
                         const float* __restrict__ Wu, const float* __restrict__ bu,
                         float* __restrict__ out_glob, float* __restrict__ gupd) {
  __shared__ float gnew[GG];
  int b = blockIdx.x, t = threadIdx.x;
  float acc = bg[t];
  const float* wr = Wg + (size_t)t * (CC + GG);
  for (int k = 0; k < CC; ++k) acc += wr[k] * agg[b * CC + k];
  for (int k = 0; k < GG; ++k) acc += wr[CC + k] * gf[b * GG + k];
  float v = tanhf(acc);
  gnew[t] = v;
  out_glob[b * GG + t] = v;
  __syncthreads();
  if (t < UU) {
    float a2 = bu[t];
    const float* ur = Wu + (size_t)t * GG;
    for (int k = 0; k < GG; ++k) a2 += ur[k] * gnew[k];
    gupd[b * UU + t] = a2;
  }
}

// ============================================================
// 4) broadcast global_update into comb[:,512:640]
// ============================================================
__global__ void k_fillgu(const float* __restrict__ gupd, _Float16* __restrict__ comb) {
  int row = blockIdx.x;            // b*NN + n
  int b = row >> 11;
  int u = threadIdx.x;
  comb[(size_t)row * KCMB + 512 + u] = (_Float16)gupd[b * UU + u];
}

// ============================================================
// WMMA helpers: per-lane A/B fragment load (16-bit, 16x16x32)
// lanes 0..15 : K = {k0..k0+7, k0+16..k0+23}
// lanes 16..31: offset +8 (handled by caller via ksel)
// ============================================================
__device__ __forceinline__ v16h load_frag(const _Float16* base, int k0) {
  v8h lo = *(const v8h*)(base + k0);
  v8h hi = *(const v8h*)(base + k0 + 16);
  v16h r;
#pragma unroll
  for (int i = 0; i < 8; ++i) { r[i] = lo[i]; r[i + 8] = hi[i]; }
  return r;
}

#define WMMA_F16(A, B, Cacc) \
  __builtin_amdgcn_wmma_f32_16x16x32_f16(false, (A), false, (B), (short)0, (Cacc), false, false)

// ============================================================
// 5) GEMM1: h = relu(x @ W_msg.T + b_msg)   [8192 x 256] x [256 -> 256]
//    wave computes a 32(M) x 32(N) register tile (2x2 accumulators);
//    block = 8 waves (4 M-waves x 2 N-waves) = 128(M) x 64(N)
// ============================================================
__global__ void k_gemm_h(const _Float16* __restrict__ comb,
                         const _Float16* __restrict__ Wm,
                         const float* __restrict__ bmsg,
                         _Float16* __restrict__ h) {
  int lane = threadIdx.x & 31;
  int wave = threadIdx.x >> 5;
  int m0 = blockIdx.y * 128 + (wave & 3) * 32;
  int n0 = blockIdx.x * 64 + (wave >> 2) * 32;
  int l15 = lane & 15;
  int ksel = (lane < 16) ? 0 : 8;
  const _Float16* arow0 = comb + (size_t)(m0 + l15) * KCMB + 256;  // x slice
  const _Float16* arow1 = arow0 + (size_t)16 * KCMB;
  const _Float16* brow0 = Wm + (size_t)(n0 + l15) * CC;            // col n of B = row n of W_msg
  const _Float16* brow1 = brow0 + (size_t)16 * CC;
  v8f c00 = {0.f,0.f,0.f,0.f,0.f,0.f,0.f,0.f};
  v8f c01 = c00, c10 = c00, c11 = c00;
#pragma unroll
  for (int kk = 0; kk < CC; kk += 32) {
    v16h a0 = load_frag(arow0, kk + ksel);
    v16h a1 = load_frag(arow1, kk + ksel);
    v16h b0 = load_frag(brow0, kk + ksel);
    v16h b1 = load_frag(brow1, kk + ksel);
    if (kk + 32 < CC) {
      // WGP-scope prefetch of next K-chunk (high locality -> near cache)
      __builtin_prefetch(arow0 + kk + 32 + ksel, 0, 3);
      __builtin_prefetch(arow1 + kk + 32 + ksel, 0, 3);
      __builtin_prefetch(brow0 + kk + 32 + ksel, 0, 3);
      __builtin_prefetch(brow1 + kk + 32 + ksel, 0, 3);
    }
    c00 = WMMA_F16(a0, b0, c00);
    c01 = WMMA_F16(a0, b1, c01);
    c10 = WMMA_F16(a1, b0, c10);
    c11 = WMMA_F16(a1, b1, c11);
  }
  int n_lo = n0 + l15, n_hi = n_lo + 16;
  float bias_lo = bmsg[n_lo], bias_hi = bmsg[n_hi];
  int mb = m0 + ((lane >> 4) << 3);
#pragma unroll
  for (int v = 0; v < 8; ++v) {
    float v00 = c00[v] + bias_lo;  v00 = v00 > 0.f ? v00 : 0.f;
    float v01 = c01[v] + bias_hi;  v01 = v01 > 0.f ? v01 : 0.f;
    float v10 = c10[v] + bias_lo;  v10 = v10 > 0.f ? v10 : 0.f;
    float v11 = c11[v] + bias_hi;  v11 = v11 > 0.f ? v11 : 0.f;
    h[(size_t)(mb + v) * HH + n_lo]      = (_Float16)v00;
    h[(size_t)(mb + v) * HH + n_hi]      = (_Float16)v01;
    h[(size_t)(mb + 16 + v) * HH + n_lo] = (_Float16)v10;
    h[(size_t)(mb + 16 + v) * HH + n_hi] = (_Float16)v11;
  }
}

// ============================================================
// 6) radius-graph neighbor mean: comb[:,0:256] = mean_{j in N(i)} h[j]
//    one block per node; 2048-bit mask built in LDS, then bit-scan gather
// ============================================================
__global__ void k_gather(const float4* __restrict__ pos4,
                         const _Float16* __restrict__ h,
                         _Float16* __restrict__ comb) {
  __shared__ unsigned int maskw[NN / 32];
  __shared__ int cnt;
  int row = blockIdx.x;            // b*NN + i
  int b = row >> 11, i = row & (NN - 1);
  int t = threadIdx.x;
  if (t == 0) cnt = 0;
  __syncthreads();

  float4 pi = pos4[row];
  unsigned int bits = 0;
  int c = 0;
#pragma unroll
  for (int k = 0; k < 8; ++k) {
    int j = t * 8 + k;
    float4 pj = pos4[(b << 11) + j];
    float dx = pi.x - pj.x, dy = pi.y - pj.y, dz = pi.z - pj.z;
    float d2 = dx * dx + dy * dy + dz * dz;
    if (d2 <= R2 && j != i) { bits |= (1u << k); ++c; }
  }
  ((unsigned char*)maskw)[t] = (unsigned char)bits;
  if (c) atomicAdd(&cnt, c);
  __syncthreads();

  float acc = 0.0f;
  const _Float16* hb = h + ((size_t)(b << 11) << 8) + t;  // h[b*NN + j][t]
#pragma unroll 4
  for (int w = 0; w < NN / 32; ++w) {
    unsigned int m = maskw[w];
    while (m) {
      int bit = __builtin_ctz(m);
      m &= m - 1;
      acc += (float)hb[(size_t)(w * 32 + bit) * HH];
    }
  }
  float den = (float)(cnt > 0 ? cnt : 1);
  comb[(size_t)row * KCMB + t] = (_Float16)(acc / den);
}

// ============================================================
// 7) GEMM2: out = comb @ W_out.T + b_out; fused transpose + position add
//    same 32x32-per-wave tiling, K = 640
// ============================================================
__global__ void k_gemm_out(const _Float16* __restrict__ comb,
                           const _Float16* __restrict__ Wo,
                           const float* __restrict__ bout,
                           const float* __restrict__ positions,
                           float* __restrict__ dout) {
  int lane = threadIdx.x & 31;
  int wave = threadIdx.x >> 5;
  int m0 = blockIdx.y * 128 + (wave & 3) * 32;
  int n0 = blockIdx.x * 64 + (wave >> 2) * 32;
  int l15 = lane & 15;
  int ksel = (lane < 16) ? 0 : 8;
  const _Float16* arow0 = comb + (size_t)(m0 + l15) * KCMB;
  const _Float16* arow1 = arow0 + (size_t)16 * KCMB;
  const _Float16* brow0 = Wo + (size_t)(n0 + l15) * KCMB;
  const _Float16* brow1 = brow0 + (size_t)16 * KCMB;
  v8f c00 = {0.f,0.f,0.f,0.f,0.f,0.f,0.f,0.f};
  v8f c01 = c00, c10 = c00, c11 = c00;
#pragma unroll 5
  for (int kk = 0; kk < KCMB; kk += 32) {
    v16h a0 = load_frag(arow0, kk + ksel);
    v16h a1 = load_frag(arow1, kk + ksel);
    v16h b0 = load_frag(brow0, kk + ksel);
    v16h b1 = load_frag(brow1, kk + ksel);
    if (kk + 32 < KCMB) {
      __builtin_prefetch(arow0 + kk + 32 + ksel, 0, 3);
      __builtin_prefetch(arow1 + kk + 32 + ksel, 0, 3);
      __builtin_prefetch(brow0 + kk + 32 + ksel, 0, 3);
      __builtin_prefetch(brow1 + kk + 32 + ksel, 0, 3);
    }
    c00 = WMMA_F16(a0, b0, c00);
    c01 = WMMA_F16(a0, b1, c01);
    c10 = WMMA_F16(a1, b0, c10);
    c11 = WMMA_F16(a1, b1, c11);
  }
  int n_lo = n0 + l15, n_hi = n_lo + 16;
  float bias_lo = bout[n_lo], bias_hi = bout[n_hi];
  int mb = m0 + ((lane >> 4) << 3);
#pragma unroll
  for (int v = 0; v < 8; ++v) {
#pragma unroll
    for (int mt = 0; mt < 2; ++mt) {
      int m = mb + mt * 16 + v;      // global row = b*NN + i
      int b = m >> 11, i = m & (NN - 1);
      float v_lo = (mt ? c10[v] : c00[v]) + bias_lo;
      float v_hi = (mt ? c11[v] : c01[v]) + bias_hi;
      // low N-tile channel
      if (n_lo < 3) {
        size_t idx = ((size_t)b * 3 + n_lo) * NN + i;
        dout[OUT_POS + idx] = positions[idx] + v_lo;
      } else {
        dout[OUT_FEAT + ((size_t)b * FF + (n_lo - 3)) * NN + i] = v_lo;
      }
      // high N-tile channel (n_hi >= 16, always a feature channel)
      dout[OUT_FEAT + ((size_t)b * FF + (n_hi - 3)) * NN + i] = v_hi;
    }
  }
}

// ============================================================
extern "C" void kernel_launch(void* const* d_in, const int* in_sizes, int n_in,
                              void* d_out, int out_size, void* d_ws, size_t ws_size,
                              hipStream_t stream) {
  const float* positions = (const float*)d_in[0];
  const float* features  = (const float*)d_in[1];
  const float* gfeat     = (const float*)d_in[2];
  const float* W_g       = (const float*)d_in[3];
  const float* b_g       = (const float*)d_in[4];
  const float* W_u       = (const float*)d_in[5];
  const float* b_u       = (const float*)d_in[6];
  const float* W_msg     = (const float*)d_in[7];
  const float* b_msg     = (const float*)d_in[8];
  const float* W_out     = (const float*)d_in[9];
  const float* b_out     = (const float*)d_in[10];
  float* out = (float*)d_out;

  char* ws = (char*)d_ws;
  float4*   pos4   = (float4*)(ws + OFF_POS4);
  _Float16* Wmsg_h = (_Float16*)(ws + OFF_WMSG);
  _Float16* Wout_h = (_Float16*)(ws + OFF_WOUT);
  float*    agg    = (float*)(ws + OFF_AGG);
  float*    gupd   = (float*)(ws + OFF_GUPD);
  _Float16* hbuf   = (_Float16*)(ws + OFF_H);
  _Float16* comb   = (_Float16*)(ws + OFF_COMB);

  // 0) pack inputs (f16 x-slice of comb) + pos4
  k_pack<<<dim3(BB * NN), 256, 0, stream>>>(positions, features, comb, pos4);
  // 1) channel max-pool
  k_maxpool<<<dim3(BB * CC), 256, 0, stream>>>(positions, features, agg);
  // 2) weight conversion (65536 + 163840 = 229376 elements)
  k_cvtw<<<dim3((HH * CC + CC * KCMB) / 256), 256, 0, stream>>>(W_msg, W_out, Wmsg_h, Wout_h);
  // 3) global MLP -> global_new (output 3) + global_update
  k_global<<<dim3(BB), 256, 0, stream>>>(agg, gfeat, W_g, b_g, W_u, b_u,
                                         out + OUT_GLOB, gupd);
  // 4) broadcast gu into comb[:,512:640]
  k_fillgu<<<dim3(BB * NN), 128, 0, stream>>>(gupd, comb);
  // 5) h = relu(x @ Wmsg.T + b) via WMMA, 128x64 per block
  k_gemm_h<<<dim3(HH / 64, (BB * NN) / 128), 256, 0, stream>>>(comb, Wmsg_h, b_msg, hbuf);
  // 6) radius-mask neighbor mean -> comb[:,0:256]
  k_gather<<<dim3(BB * NN), 256, 0, stream>>>(pos4, hbuf, comb);
  // 7) out = comb @ Wout.T + b, fused transpose + position residual
  k_gemm_out<<<dim3(CC / 64, (BB * NN) / 128), 256, 0, stream>>>(comb, Wout_h, b_out,
                                                                 positions, out);
}
